// attention_11269994185437
// MI455X (gfx1250) — matrix operand order repaired and checked by reading, not compiled, not run on hardware
//
#include <hip/hip_runtime.h>
#include <hip/hip_bf16.h>
#include <cstdint>

// ---------------------------------------------------------------------------
// Problem constants
// ---------------------------------------------------------------------------
#define BATCH 4
#define CCH   256          // channels
#define CQK   32           // q/k channels
#define NPIX  4096         // H*W
#define MTOT  320          // CQK + CQK + CCH rows of fused weight

// Workspace layout (bytes, all 256-aligned). Total ~153.4 MB.
#define OFF_WBF   ((size_t)0)                       // 320*256 bf16      = 163840
#define OFF_BIAS  ((size_t)163840)                  // 320 f32           = 1280
#define OFF_XT    ((size_t)165376)                  // 4*4096*256 bf16   = 8388608
#define OFF_QT    ((size_t)8553984)                 // 4*4096*32 bf16    = 1048576
#define OFF_KT    ((size_t)9602560)                 // 1048576
#define OFF_V     ((size_t)10651136)                // 4*256*4096 bf16   = 8388608
#define OFF_RMAX  ((size_t)19039744)                // 4*4096 f32        = 65536
#define OFF_RRCP  ((size_t)19105280)                // 65536
#define OFF_AT    ((size_t)19170816)                // 4*4096*4096 bf16  = 134217728 (L2-resident)

typedef __attribute__((ext_vector_type(16))) __bf16 v16bf;
typedef __attribute__((ext_vector_type(8)))  __bf16 v8bf;
typedef __attribute__((ext_vector_type(8)))  float  v8f;

__device__ __forceinline__ v8f vzero8() {
    v8f z;
#pragma unroll
    for (int i = 0; i < 8; ++i) z[i] = 0.0f;
    return z;
}

// ---------------------------------------------------------------------------
// gfx1250 async global->LDS copy (ASYNCcnt-tracked, writes LDS directly,
// no VGPR staging).  GV mode: vdst = per-lane LDS byte address, vaddr =
// per-lane 64-bit global address.
// ---------------------------------------------------------------------------
__device__ __forceinline__ void async_load_b128(const void* gaddr, uint32_t lds_byte) {
    asm volatile("global_load_async_to_lds_b128 %0, %1, off"
                 :: "v"(lds_byte), "v"(gaddr)
                 : "memory");
}
#define WAIT_ASYNCCNT(n) asm volatile("s_wait_asynccnt %0" :: "i"(n) : "memory")

__device__ __forceinline__ uint32_t lds_u32(const void* p) {
    // AS(3) objects: low 32 bits of the generic pointer are the LDS address.
    return (uint32_t)(uintptr_t)p;
}

// ---------------------------------------------------------------------------
// WMMA fragment loaders.
// A-matrix 16x32 bf16 (row-major [M,K], ldk elems):
//   lanes 0-15  : row M=lane,   regs hold K{0..7},{16..23}
//   lanes 16-31 : row M=lane-16,regs hold K{8..15},{24..31}
// B-matrix 32x16 bf16 loaded from B^T row-major [N,K]:
//   lanes 0-15  : col N=lane,   K{0..7},{16..23}
//   lanes 16-31 : col N=lane-16,K{8..15},{24..31}
// Each fragment = two 16-byte loads per lane (ds_load_b128 / global_load_b128).
// ---------------------------------------------------------------------------
__device__ __forceinline__ v16bf load_frag_a(const __bf16* base, int ldk,
                                             int row, int k0, int khalf) {
    const __bf16* p = base + (size_t)row * ldk + k0;
    v8bf lo = *(const v8bf*)(p + khalf * 8);
    v8bf hi = *(const v8bf*)(p + khalf * 8 + 16);
    v16bf r;
#pragma unroll
    for (int i = 0; i < 8; ++i) { r[i] = lo[i]; r[i + 8] = hi[i]; }
    return r;
}

__device__ __forceinline__ v16bf load_frag_b(const __bf16* base, int ldk,
                                             int row, int k0, int khalf) {
    const __bf16* p = base + (size_t)row * ldk + k0 + khalf * 8;
    v8bf lo = *(const v8bf*)(p);
    v8bf hi = *(const v8bf*)(p + 16);
    v16bf r;
#pragma unroll
    for (int i = 0; i < 8; ++i) { r[i] = lo[i]; r[i + 8] = hi[i]; }
    return r;
}

__device__ __forceinline__ v8f wmma_bf16(v16bf a, v16bf b, v8f c) {
    return __builtin_amdgcn_wmma_f32_16x16x32_bf16(
        /*neg_a=*/false, a, /*neg_b=*/false, b,
        /*c_mod=*/(short)0, c, /*reuse_a=*/false, /*reuse_b=*/false);
}

// ---------------------------------------------------------------------------
// Kernel 0a: fuse Wq|Wk|Wv -> bf16 [320,256], biases -> f32 [320]
// ---------------------------------------------------------------------------
__global__ void k_pack_w(const float* __restrict__ Wq, const float* __restrict__ bq,
                         const float* __restrict__ Wk, const float* __restrict__ bk,
                         const float* __restrict__ Wv, const float* __restrict__ bv,
                         __bf16* __restrict__ Wbf, float* __restrict__ bias) {
    int row = blockIdx.x;       // 0..319
    int c   = threadIdx.x;      // 0..255
    const float* src; const float* bs; int r;
    if (row < 32)      { src = Wq; bs = bq; r = row; }
    else if (row < 64) { src = Wk; bs = bk; r = row - 32; }
    else               { src = Wv; bs = bv; r = row - 64; }
    Wbf[(size_t)row * CCH + c] = (__bf16)src[(size_t)r * CCH + c];
    if (c == 0) bias[row] = bs[r];
}

// ---------------------------------------------------------------------------
// Kernel 0b: x [B,C,N] f32 -> xT [B,N,C] bf16 (tiled transpose via LDS)
// grid (N/32, C/32, B), block (32,8)
// ---------------------------------------------------------------------------
__global__ void k_xpose(const float* __restrict__ x, __bf16* __restrict__ xT) {
    int b  = blockIdx.z;
    int c0 = blockIdx.y * 32;
    int n0 = blockIdx.x * 32;
    __shared__ float tile[32][33];
#pragma unroll
    for (int i = 0; i < 4; ++i) {
        int cl = threadIdx.y + 8 * i;
        tile[cl][threadIdx.x] =
            x[((size_t)b * CCH + c0 + cl) * NPIX + n0 + threadIdx.x];
    }
    __syncthreads();
#pragma unroll
    for (int i = 0; i < 4; ++i) {
        int nl = threadIdx.y + 8 * i;
        xT[((size_t)b * NPIX + n0 + nl) * CCH + c0 + threadIdx.x] =
            (__bf16)tile[threadIdx.x][nl];
    }
}

// ---------------------------------------------------------------------------
// Kernel 1: fused QKV GEMM.  [320,256] x [256,4096] per batch, WMMA bf16.
// Outputs: qT [B,N,32], kT [B,N,32] (A/B^T layouts for the logits GEMM),
//          v [B,C,N] bf16 (A layout for the out GEMM). Biases folded in.
// grid (64, 5, B), block 128 (4 waves, 64x64 tile)
// ---------------------------------------------------------------------------
__global__ __launch_bounds__(128) void k_qkv(const __bf16* __restrict__ Wbf,
                                             const float* __restrict__ bias,
                                             const __bf16* __restrict__ xT,
                                             __bf16* __restrict__ qT,
                                             __bf16* __restrict__ kT,
                                             __bf16* __restrict__ v) {
    int b  = blockIdx.z;
    int m0 = blockIdx.y * 64;
    int n0 = blockIdx.x * 64;
    int t = threadIdx.x, lane = t & 31, w = t >> 5;
    int mrow = lane & 15, khalf = lane >> 4;

    __shared__ __align__(16) __bf16 Wt[64 * 128];
    __shared__ __align__(16) __bf16 Xt[64 * 128];

    v8f acc[4];
#pragma unroll
    for (int j = 0; j < 4; ++j) acc[j] = vzero8();

    for (int kc = 0; kc < CCH; kc += 128) {
        const __bf16* Wg = Wbf + (size_t)m0 * CCH + kc;
        const __bf16* Xg = xT + ((size_t)b * NPIX + n0) * CCH + kc;
#pragma unroll
        for (int i = 0; i < 8; ++i) {               // 64x128 bf16 per tile
            int idx = t + 128 * i;                  // 1024 x 8-elem chunks
            int row = idx >> 4;
            int c8  = (idx & 15) * 8;
            *(v8bf*)&Wt[row * 128 + c8] = *(const v8bf*)&Wg[(size_t)row * CCH + c8];
            *(v8bf*)&Xt[row * 128 + c8] = *(const v8bf*)&Xg[(size_t)row * CCH + c8];
        }
        __syncthreads();
#pragma unroll
        for (int kk = 0; kk < 4; ++kk) {
            int k0 = kk * 32;
            v16bf a = load_frag_a(Wt, 128, 16 * w + mrow, k0, khalf);
#pragma unroll
            for (int j = 0; j < 4; ++j) {
                v16bf bb = load_frag_b(Xt, 128, 16 * j + mrow, k0, khalf);
                acc[j] = wmma_bf16(a, bb, acc[j]);
            }
        }
        __syncthreads();
    }
    // Epilogue: add bias, route q / k (transposed) / v.
#pragma unroll
    for (int j = 0; j < 4; ++j) {
        int n = n0 + 16 * j + mrow;
#pragma unroll
        for (int r = 0; r < 8; ++r) {
            int m = m0 + 16 * w + r + 8 * khalf;
            float val = acc[j][r] + bias[m];
            __bf16 h = (__bf16)val;
            if (m < CQK)            qT[((size_t)b * NPIX + n) * CQK + m] = h;
            else if (m < 2 * CQK)   kT[((size_t)b * NPIX + n) * CQK + (m - CQK)] = h;
            else                    v[((size_t)b * CCH + (m - 2 * CQK)) * NPIX + n] = h;
        }
    }
}

// ---------------------------------------------------------------------------
// Kernel 2a: softmax row stats (pass A). S tile = WMMA(qT, kT) with K=32.
// Per-lane online (max,sum), combined across 16-lane halves via shfl_xor.
// grid (64, B), block 128; wave w owns rows n0+16w..+15.
// ---------------------------------------------------------------------------
__global__ __launch_bounds__(128) void k_rowstats(const __bf16* __restrict__ qT,
                                                  const __bf16* __restrict__ kT,
                                                  float* __restrict__ rmax,
                                                  float* __restrict__ rrcp) {
    int b = blockIdx.y;
    int n0 = blockIdx.x * 64;
    int t = threadIdx.x, lane = t & 31, w = t >> 5;
    int mrow = lane & 15, khalf = lane >> 4;

    const __bf16* q = qT + (size_t)b * NPIX * CQK;
    const __bf16* k = kT + (size_t)b * NPIX * CQK;

    v16bf a = load_frag_a(q, CQK, n0 + 16 * w + mrow, 0, khalf);

    float mx[8], sm[8];
#pragma unroll
    for (int r = 0; r < 8; ++r) { mx[r] = -1e30f; sm[r] = 0.0f; }

    for (int mt = 0; mt < NPIX / 16; ++mt) {
        v16bf bb = load_frag_b(k, CQK, mt * 16 + mrow, 0, khalf);
        v8f s = wmma_bf16(a, bb, vzero8());
#pragma unroll
        for (int r = 0; r < 8; ++r) {
            float xv = s[r];
            float nm = fmaxf(mx[r], xv);
            sm[r] = sm[r] * __expf(mx[r] - nm) + __expf(xv - nm);
            mx[r] = nm;
        }
    }
#pragma unroll
    for (int r = 0; r < 8; ++r) {
        float m_ = mx[r], s_ = sm[r];
#pragma unroll
        for (int o = 8; o > 0; o >>= 1) {          // stay inside 16-lane half
            float om = __shfl_xor(m_, o, 32);
            float os = __shfl_xor(s_, o, 32);
            float nm = fmaxf(m_, om);
            s_ = s_ * __expf(m_ - nm) + os * __expf(om - nm);
            m_ = nm;
        }
        if (mrow == 0) {
            int n = n0 + 16 * w + r + 8 * khalf;
            rmax[(size_t)b * NPIX + n] = m_;
            rrcp[(size_t)b * NPIX + n] = 1.0f / s_;
        }
    }
}

// ---------------------------------------------------------------------------
// Kernel 2b: recompute S, apply exp/scale, write A^T bf16 [B, m, n]
// (B^T layout for the out-GEMM). Block tile 64m x 64n, staged via LDS for
// 16B-chunk writeout. grid (64 mblk, 64 nblk, B), block 128.
// ---------------------------------------------------------------------------
__global__ __launch_bounds__(128) void k_attn(const __bf16* __restrict__ qT,
                                              const __bf16* __restrict__ kT,
                                              const float* __restrict__ rmax,
                                              const float* __restrict__ rrcp,
                                              __bf16* __restrict__ At) {
    int b  = blockIdx.z;
    int m0 = blockIdx.x * 64;
    int n0 = blockIdx.y * 64;
    int t = threadIdx.x, lane = t & 31, w = t >> 5;
    int mrow = lane & 15, khalf = lane >> 4;

    __shared__ __align__(16) __bf16 stage[64 * 72];  // [m_loc][n_loc], padded
    __shared__ float smax[64];
    __shared__ float srcp[64];

    if (t < 64) {
        smax[t] = rmax[(size_t)b * NPIX + n0 + t];
        srcp[t] = rrcp[(size_t)b * NPIX + n0 + t];
    }
    __syncthreads();

    const __bf16* q = qT + (size_t)b * NPIX * CQK;
    const __bf16* k = kT + (size_t)b * NPIX * CQK;

    v16bf a = load_frag_a(q, CQK, n0 + 16 * w + mrow, 0, khalf);
#pragma unroll
    for (int j = 0; j < 4; ++j) {
        v16bf bb = load_frag_b(k, CQK, m0 + 16 * j + mrow, 0, khalf);
        v8f s = wmma_bf16(a, bb, vzero8());
        int ml = 16 * j + mrow;
#pragma unroll
        for (int r = 0; r < 8; ++r) {
            int nl = 16 * w + r + 8 * khalf;
            float p = __expf(s[r] - smax[nl]) * srcp[nl];
            stage[ml * 72 + nl] = (__bf16)p;
        }
    }
    __syncthreads();
    // 64 rows x 128B -> 512 16-byte chunks, 128 threads x 4
#pragma unroll
    for (int i = 0; i < 4; ++i) {
        int idx = t + 128 * i;
        int row = idx >> 3;
        int c8  = (idx & 7) * 8;
        *(v8bf*)&At[((size_t)b * NPIX + m0 + row) * NPIX + n0 + c8] =
            *(const v8bf*)&stage[row * 72 + c8];
    }
}

// ---------------------------------------------------------------------------
// Kernel 3: out = V @ A, then d_out = gamma*out + x.
// M=256, N=4096, K=4096 per batch; A-op = v [C,N], B-op = At [m,n].
// Double-buffered GLOBAL_LOAD_ASYNC_TO_LDS_B128 staging (ASYNCcnt-tracked):
// while WMMAs consume slab k, the 4 async b128s/thread for slab k+1 stream
// L2 -> LDS with no VGPR staging.  s_wait_asynccnt 4 at the top of a step
// retires the previous slab's loads (async loads complete in order); the
// end-of-step barrier protects the buffer being refilled.
// grid (64 Ntile, 4 Mtile, B), block 128.
// ---------------------------------------------------------------------------
__global__ __launch_bounds__(128) void k_out(const __bf16* __restrict__ v,
                                             const __bf16* __restrict__ At,
                                             const float* __restrict__ x,
                                             const float* __restrict__ gamma,
                                             float* __restrict__ out) {
    int b  = blockIdx.z;
    int c0 = blockIdx.y * 64;
    int m0 = blockIdx.x * 64;
    int t = threadIdx.x, lane = t & 31, w = t >> 5;
    int mrow = lane & 15, khalf = lane >> 4;

    __shared__ __align__(16) __bf16 Vt[2][64 * 40];   // padded ldk=40 (16B mult)
    __shared__ __align__(16) __bf16 Bt[2][64 * 40];

    v8f acc[4];
#pragma unroll
    for (int j = 0; j < 4; ++j) acc[j] = vzero8();

    const __bf16* Vg0 = v + ((size_t)b * CCH + c0) * NPIX;
    const __bf16* Ag0 = At + ((size_t)b * NPIX + m0) * NPIX;

    // this thread's two 8-elem chunks of each 64x32 tile
    int row0 = t >> 2, c80 = (t & 3) * 8;             // idx = t
    int row1 = (t + 128) >> 2, c81 = ((t + 128) & 3) * 8;

    auto issue = [&](int buf, int k0g) {
        async_load_b128(&Vg0[(size_t)row0 * NPIX + k0g + c80],
                        lds_u32(&Vt[buf][row0 * 40 + c80]));
        async_load_b128(&Ag0[(size_t)row0 * NPIX + k0g + c80],
                        lds_u32(&Bt[buf][row0 * 40 + c80]));
        async_load_b128(&Vg0[(size_t)row1 * NPIX + k0g + c81],
                        lds_u32(&Vt[buf][row1 * 40 + c81]));
        async_load_b128(&Ag0[(size_t)row1 * NPIX + k0g + c81],
                        lds_u32(&Bt[buf][row1 * 40 + c81]));
    };

    issue(0, 0);                                      // prologue
    for (int ks = 0; ks < NPIX / 32; ++ks) {
        int cur = ks & 1;
        if (ks + 1 < NPIX / 32) {
            issue(cur ^ 1, (ks + 1) * 32);            // prefetch next slab
            WAIT_ASYNCCNT(4);                         // slab ks landed
        } else {
            WAIT_ASYNCCNT(0);
        }
        __syncthreads();                              // block-wide visibility
        v16bf a = load_frag_a(Vt[cur], 40, 16 * w + mrow, 0, khalf);
#pragma unroll
        for (int j = 0; j < 4; ++j) {
            v16bf bb = load_frag_b(Bt[cur], 40, 16 * j + mrow, 0, khalf);
            acc[j] = wmma_bf16(a, bb, acc[j]);
        }
        __syncthreads();                              // done reading buf cur
    }

    float g = gamma[0];
#pragma unroll
    for (int j = 0; j < 4; ++j) {
        int m = m0 + 16 * j + mrow;
#pragma unroll
        for (int r = 0; r < 8; ++r) {
            int c = c0 + 16 * w + r + 8 * khalf;
            size_t off = ((size_t)b * CCH + c) * NPIX + m;
            out[off] = g * acc[j][r] + x[off];
        }
    }
}

// ---------------------------------------------------------------------------
// Launch
// ---------------------------------------------------------------------------
extern "C" void kernel_launch(void* const* d_in, const int* in_sizes, int n_in,
                              void* d_out, int out_size, void* d_ws, size_t ws_size,
                              hipStream_t stream) {
    const float* x     = (const float*)d_in[0];
    const float* Wq    = (const float*)d_in[1];
    const float* bq    = (const float*)d_in[2];
    const float* Wk    = (const float*)d_in[3];
    const float* bk    = (const float*)d_in[4];
    const float* Wv    = (const float*)d_in[5];
    const float* bv    = (const float*)d_in[6];
    const float* gamma = (const float*)d_in[7];
    float* out = (float*)d_out;

    char* ws = (char*)d_ws;
    __bf16* Wbf  = (__bf16*)(ws + OFF_WBF);
    float*  bias = (float*)(ws + OFF_BIAS);
    __bf16* xT   = (__bf16*)(ws + OFF_XT);
    __bf16* qT   = (__bf16*)(ws + OFF_QT);
    __bf16* kT   = (__bf16*)(ws + OFF_KT);
    __bf16* vbf  = (__bf16*)(ws + OFF_V);
    float*  rmax = (float*)(ws + OFF_RMAX);
    float*  rrcp = (float*)(ws + OFF_RRCP);
    __bf16* At   = (__bf16*)(ws + OFF_AT);

    k_pack_w<<<dim3(MTOT), dim3(CCH), 0, stream>>>(Wq, bq, Wk, bk, Wv, bv, Wbf, bias);
    k_xpose<<<dim3(NPIX / 32, CCH / 32, BATCH), dim3(32, 8), 0, stream>>>(x, xT);
    k_qkv<<<dim3(NPIX / 64, MTOT / 64, BATCH), dim3(128), 0, stream>>>(Wbf, bias, xT, qT, kT, vbf);
    k_rowstats<<<dim3(NPIX / 64, BATCH), dim3(128), 0, stream>>>(qT, kT, rmax, rrcp);
    k_attn<<<dim3(NPIX / 64, NPIX / 64, BATCH), dim3(128), 0, stream>>>(qT, kT, rmax, rrcp, At);
    k_out<<<dim3(NPIX / 64, CCH / 64, BATCH), dim3(128), 0, stream>>>(vbf, At, x, gamma, out);
}
